// CostVolumeLayer_893353197594
// MI455X (gfx1250) — matrix operand order repaired and checked by reading, not compiled, not run on hardware
//
#include <hip/hip_runtime.h>
#include <stdint.h>

typedef __attribute__((ext_vector_type(16))) __bf16        v16bf;
typedef __attribute__((ext_vector_type(8)))  uint32_t      v8u;
typedef __attribute__((ext_vector_type(8)))  float         v8f;

#define RNG    4
#define DD     9
#define NDISP  81
#define NB     8
#define NC     192
#define NH     128
#define NW_IMG 256
#define HW     (NH * NW_IMG)
#define KSTEPS (NC / 32)          // 6 K-steps of 32 channels
#define WAVES_PER_BLOCK 8

// Split (a,b) into bf16 hi-pair and bf16 lo-pair (truncation split):
//   hi = trunc_bf16(x), lo = trunc_bf16(x - hi);  hi+lo ~ x to ~2^-16 rel.
// Lowers to v_perm_b32 + v_fma_mix_f32_bf16 (1 op each).
struct pk2 { uint32_t hi, lo; };
__device__ __forceinline__ pk2 split_pair(float a, float b) {
    pk2 r;
    const uint32_t b0 = __float_as_uint(a);
    const uint32_t b1 = __float_as_uint(b);
    r.hi = __builtin_amdgcn_perm(b1, b0, 0x07060302u);   // [b1.hi16 | b0.hi16]
    const float ha = __uint_as_float(b0 & 0xFFFF0000u);
    const float hb = __uint_as_float(b1 & 0xFFFF0000u);
    const uint32_t r0 = __float_as_uint(a - ha);
    const uint32_t r1 = __float_as_uint(b - hb);
    r.lo = __builtin_amdgcn_perm(r1, r0, 0x07060302u);
    return r;
}

// One x2 row -> one 16-wide N-tile, 6 K-steps, 18 WMMAs.
// MASK=false for interior w-tiles: no per-element cndmask.
template <bool MASK>
__device__ __forceinline__ void accumulate_row(const float* __restrict__ r0,
                                               bool ok0, int koff,
                                               const v16bf* ahi, const v16bf* alo,
                                               v8f& acc) {
    #pragma unroll
    for (int s = 0; s < KSTEPS; ++s) {
        v8u uh0, ul0;
        #pragma unroll
        for (int j = 0; j < 8; ++j) {
            const long co = (long)(s * 32 + koff + 2 * j) * HW;
            float a0 = r0[co];            // unconditional loads (clamped addr)
            float a1 = r0[co + HW];
            if (MASK) {
                a0 = ok0 ? a0 : 0.0f;     // v_cndmask, edge tiles only
                a1 = ok0 ? a1 : 0.0f;
            }
            const pk2 ra = split_pair(a0, a1);
            uh0[j] = ra.hi; ul0[j] = ra.lo;
        }
        const v16bf bh0 = __builtin_bit_cast(v16bf, uh0);
        const v16bf bl0 = __builtin_bit_cast(v16bf, ul0);
        // split product: x1*x2 ~ hi*hi + hi*lo + lo*hi, fp32 accumulate
        acc = __builtin_amdgcn_wmma_f32_16x16x32_bf16(false, ahi[s], false, bh0, (short)0, acc, false, false);
        acc = __builtin_amdgcn_wmma_f32_16x16x32_bf16(false, ahi[s], false, bl0, (short)0, acc, false, false);
        acc = __builtin_amdgcn_wmma_f32_16x16x32_bf16(false, alo[s], false, bh0, (short)0, acc, false, false);
    }
}

__global__ __launch_bounds__(256)
__attribute__((amdgpu_waves_per_eu(4)))      // cap <=256 VGPRs
void CostVolumeLayer_893353197594_kernel(const float* __restrict__ x1,
                                         const float* __restrict__ x2,
                                         float* __restrict__ out) {
    __shared__ float G[WAVES_PER_BLOCK][16][17];   // 16x16 Gram tile per wave (+pad)

    const int lane  = threadIdx.x & 31;
    const int wv    = threadIdx.x >> 5;
    const int gwave = blockIdx.x * WAVES_PER_BLOCK + wv;

    const int wt = gwave & 31;            // 32 w-tiles of 8 pixels
    const int hp = (gwave >> 5) & 63;     // 64 h-pairs
    const int b  = gwave >> 11;           // 8 batches
    const int w0 = wt * 8;
    const int h  = hp * 2;

    const int half = lane >> 4;           // lane half selects K interleave
    const int mn   = lane & 15;           // M (A rows) / N (B cols) index

    // M-tile = 2 rows x 8 cols: M index m -> pixel (h + (m>>3), w0 + (m&7)).
    // All 9 j-displacements of an 8-wide pixel strip fit one 16-wide N-tile.
    const int arow = mn >> 3;             // 0/1: x1 row within pair
    const int acol = mn & 7;              // x1 column within 8-strip

    // ---- A fragments: x1[b, :, h+arow, w0+acol], split hi/lo, cached for all rows.
    // 16x32 bf16 A layout (matches B): lanes0-15 K=0..15, lanes16-31 K=16..31
    v16bf ahi[KSTEPS], alo[KSTEPS];
    {
        const float* __restrict__ x1p =
            x1 + ((long)(b * NC) * NH + (h + arow)) * NW_IMG + (w0 + acol);
        #pragma unroll
        for (int s = 0; s < KSTEPS; ++s) {
            v8u uh, ul;
            #pragma unroll
            for (int j = 0; j < 8; ++j) {          // pairs (t=2j,2j+1) never straddle t=8
                const int t0 = 2 * j;
                const int k0 = s * 32 + t0 + 16 * half;
                const float f0 = x1p[(long)k0 * HW];
                const float f1 = x1p[(long)(k0 + 1) * HW];
                const pk2 r = split_pair(f0, f1);
                uh[j] = r.hi;
                ul[j] = r.lo;
            }
            ahi[s] = __builtin_bit_cast(v16bf, uh);
            alo[s] = __builtin_bit_cast(v16bf, ul);
        }
    }

    // B-frag 32x16 bf16 layout: lanes0-15 -> N=lane, K=kb+0..15; lanes16-31 -> N=lane-16, K=kb+16..31
    const int  koff = half * 16;
    const int  p0   = w0 - 4 + mn;                 // x2 column for this lane, [w0-4, w0+11]
    const bool ok0  = (p0 >= 0) && (p0 < NW_IMG);
    const int  p0c  = p0 < 0 ? 0 : (p0 >= NW_IMG ? NW_IMG - 1 : p0);
    const bool edge = (w0 == 0) || (w0 == NW_IMG - 8);

    const long x2batch = (long)(b * NC) * NH * NW_IMG;

    // x2 rows hh = h-4 .. h+5: row hh serves x1-row h (ii=hh-h) and h+1 (ii=hh-h-1)
    for (int q = 0; q < 10; ++q) {
        const int hh = h - 4 + q;
        v8f acc = {};                      // 16x16 G tile: rows = M (2x8 pixels), cols = x2 cols
        if (hh >= 0 && hh < NH) {
            const float* __restrict__ r0 = x2 + x2batch + (long)hh * NW_IMG + p0c;
            if (hh + 1 < NH) __builtin_prefetch(r0 + NW_IMG, 0, 3);
            if (edge)
                accumulate_row<true >(r0, ok0, koff, ahi, alo, acc);
            else
                accumulate_row<false>(r0, ok0, koff, ahi, alo, acc);
        }

        // Spill G (16x16 fp32) to wave-private LDS per C/D layout:
        // VGPR r: lanes0-15 -> (M=r, N=lane), lanes16-31 -> (M=r+8, N=lane-16)
        #pragma unroll
        for (int r = 0; r < 8; ++r) {
            G[wv][r + 8 * half][mn] = acc[r];
        }
        asm volatile("s_wait_dscnt 0" ::: "memory");

        // Extraction: out pixel (h+rr, w0+cc), displacement jj = j-4 reads G[m][cc+j].
        // Row hh contributes iff ii = hh-h-rr in [-4,4]; each output written exactly once.
        #pragma unroll
        for (int t5 = 0; t5 < 5; ++t5) {
            const int t = t5 * 32 + lane;
            if (t < 144) {
                const int m  = t & 15;
                const int j  = t >> 4;             // 0..8
                const int rr = m >> 3;
                const int cc = m & 7;
                const int iir = (q - 4) - rr;      // = hh - h - rr
                if (iir >= -RNG && iir <= RNG) {
                    const float g = G[wv][m][cc + j] * (1.0f / 81.0f);
                    // reference: chans[(9*i + jref) % 81] uses x2 shifted by (-i, -j)
                    const int iref = -iir;
                    const int jref = 4 - j;
                    int idx = (DD * iref + jref) % NDISP;
                    if (idx < 0) idx += NDISP;
                    out[(((long)b * NDISP + idx) * NH + (h + rr)) * NW_IMG + (w0 + cc)] = g;
                }
            }
        }
        asm volatile("s_wait_dscnt 0" ::: "memory");  // reads done before next q overwrites G
    }
}

extern "C" void kernel_launch(void* const* d_in, const int* in_sizes, int n_in,
                              void* d_out, int out_size, void* d_ws, size_t ws_size,
                              hipStream_t stream) {
    (void)in_sizes; (void)n_in; (void)d_ws; (void)ws_size; (void)out_size;
    const float* x1 = (const float*)d_in[0];
    const float* x2 = (const float*)d_in[1];
    float* out = (float*)d_out;

    const int total_waves = NB * (NH / 2) * (NW_IMG / 8);   // 16384 waves, one per (b, h-pair, 8-wide w-tile)
    const int blocks = total_waves / WAVES_PER_BLOCK;       // 2048 blocks of 256 threads
    CostVolumeLayer_893353197594_kernel<<<blocks, 256, 0, stream>>>(x1, x2, out);
}
